// DecoderBlock_51367808860309
// MI455X (gfx1250) — matrix operand order repaired and checked by reading, not compiled, not run on hardware
//
#include <hip/hip_runtime.h>
#include <math.h>

typedef __attribute__((ext_vector_type(16))) __bf16 v16bf;
typedef __attribute__((ext_vector_type(8)))  __bf16 v8bf;
typedef __attribute__((ext_vector_type(8)))  float  v8f;
typedef __attribute__((ext_vector_type(4)))  float  v4f;
typedef __attribute__((ext_vector_type(4)))  int    v4i;

#define BZ   256
#define SEQ  512
#define INSZ 256
#define HID  256
#define M_TILES ((BZ * SEQ) / 16)   // 8192 row-tiles for the XU GEMM

// Async global->LDS staging (gfx1250 ASYNCcnt path), guarded so the file
// compiles identically on toolchains without the builtin.
#if defined(__HIP_DEVICE_COMPILE__) && __has_builtin(__builtin_amdgcn_global_load_async_to_lds_b128)
#define USE_ASYNC_STAGE 1
#else
#define USE_ASYNC_STAGE 0
#endif

// ---------------------------------------------------------------- helpers ---

__device__ __forceinline__ float fast_tanh(float x) {
#if __has_builtin(__builtin_amdgcn_tanhf)
  return __builtin_amdgcn_tanhf(x);
#else
  return tanhf(x);
#endif
}

__device__ __forceinline__ v16bf join8(v8bf lo, v8bf hi) {
  return __builtin_shufflevector(lo, hi, 0,1,2,3,4,5,6,7,8,9,10,11,12,13,14,15);
}

__device__ __forceinline__ v8bf cvt8(v4f a, v4f b) {
  v8bf o;
  o[0] = (__bf16)a[0]; o[1] = (__bf16)a[1]; o[2] = (__bf16)a[2]; o[3] = (__bf16)a[3];
  o[4] = (__bf16)b[0]; o[5] = (__bf16)b[1]; o[6] = (__bf16)b[2]; o[7] = (__bf16)b[3];
  return o;
}

// A fragment (16x32 bf16, ISA 7.12.2 layout) from a row-major bf16 LDS tile
// [16][256]. lane L: M = L%16 ; K = kb*32 + (L/16)*8 + {0..7} and +16.
__device__ __forceinline__ v16bf load_a_frag(const __bf16* sA, int lane, int kb) {
  const int m  = lane & 15;
  const int gg = lane >> 4;
  const __bf16* p = sA + m * 256 + kb * 32 + gg * 8;   // byte offsets 16B-aligned
  v8bf lo = *(const v8bf*)p;
  v8bf hi = *(const v8bf*)(p + 16);
  return join8(lo, hi);
}

// B fragment (32x16 bf16) gathered straight from the global f32 weight matrix
// W[K][N] (row-major, row stride 256). Per ISA layout: lane L holds N = L%16,
// K = kb*32 + (L/16)*8 + {0..7} and +16. One-time cost; W rows stay hot in L2.
__device__ __forceinline__ v16bf load_w_frag_global(const float* __restrict__ W,
                                                    int lane, int kb, int nt) {
  const int n     = (nt << 4) + (lane & 15);
  const int kbase = (kb << 5) + ((lane >> 4) << 3);
  v16bf r;
#pragma unroll
  for (int e = 0; e < 16; ++e)
    r[e] = (__bf16)W[(size_t)(kbase + e + ((e >> 3) << 3)) * 256 + n];
  return r;
}

// Synchronous staging: 16x256 f32 -> bf16 LDS. 512 threads x 8 contiguous
// elements: 2x global_load_b128 + 4x v_cvt_pk_bf16_f32 + 1x ds_store_b128.
__device__ __forceinline__ void stage_tile(__bf16* dst,
                                           const float* __restrict__ src, int tid) {
  const int i = tid << 3;                  // 0..4088
  v4f a = *(const v4f*)(src + i);
  v4f b = *(const v4f*)(src + i + 4);
  *(v8bf*)(dst + i) = cvt8(a, b);
}

#if USE_ASYNC_STAGE
// Builtin signature (from hipcc diagnostic): (v4i addrspace(1)*, v4i addrspace(3)*,
// imm offset, imm cpol) -- offset applies to both addresses per ISA 10.7.
typedef __attribute__((address_space(1))) v4i* gv4i_p;
typedef __attribute__((address_space(3))) v4i* lv4i_p;

// Async staging: 16x256 f32 tile copied global->LDS by the async engine,
// no VGPR round-trip. 512 threads x 32 B = 2x global_load_async_to_lds_b128.
__device__ __forceinline__ void stage_tile_async(float* dstLds,
                                                 const float* __restrict__ src,
                                                 int tid) {
  const int i = tid << 3;
  gv4i_p g = (gv4i_p)(src + i);
  lv4i_p l = (lv4i_p)(dstLds + i);
  __builtin_amdgcn_global_load_async_to_lds_b128(g, l, 0, 0);
  __builtin_amdgcn_global_load_async_to_lds_b128(g, l, 16, 0);
}

// Each thread converts exactly the 8 f32 elements it async-staged, so no
// barrier is needed between the asynccnt wait and this conversion.
__device__ __forceinline__ void convert_tile(__bf16* dst, const float* srcLds,
                                             int tid) {
  const int i = tid << 3;
  v4f a = *(const v4f*)(srcLds + i);
  v4f b = *(const v4f*)(srcLds + i + 4);
  *(v8bf*)(dst + i) = cvt8(a, b);
}

__device__ __forceinline__ void wait_async0() {
#if __has_builtin(__builtin_amdgcn_s_wait_asynccnt)
  __builtin_amdgcn_s_wait_asynccnt(0);
#else
  asm volatile("s_wait_asynccnt 0x0" ::: "memory");
#endif
}
#endif

// ------------------------------------------------- phase 1: XU = x @ U -----
// 512 threads = 16 waves; wave w owns N-tile w and keeps its 8 U fragments in
// registers (64 VGPRs). LDS: 16 KB f32 async-staging buffer + double-buffered
// bf16 A tile. The async engine copies tile t+1 while tile t computes, then
// tile t+1 is converted f32->bf16 (thread-private ranges). One barrier/tile,
// and the WMMA chain keeps the low-pressure 2-load bf16 fragments (no spills).
__global__ void rnn_xu_kernel(const float* __restrict__ x,   // [BZ*SEQ, INSZ]
                              const float* __restrict__ U,   // [INSZ, HID]
                              float* __restrict__ out) {     // d_out
  extern __shared__ char smem[];
#if USE_ASYNC_STAGE
  float*  sF = (float*)smem;                  // 16 KB async staging (f32)
  __bf16* sB = (__bf16*)(smem + 16384);       // 2 x 8 KB bf16 tiles
#else
  __bf16* sB = (__bf16*)smem;                 // 2 x 8 KB bf16 tiles
#endif

  const int tid  = threadIdx.x;
  const int lane = tid & 31;
  const int wave = tid >> 5;                  // N-tile index 0..15

  // register-resident B operand (invariant per wave)
  v16bf ufrag[8];
#pragma unroll
  for (int kb = 0; kb < 8; ++kb)
    ufrag[kb] = load_w_frag_global(U, lane, kb, wave);

  float* dec = out + BZ * HID;                // decoder_outputs region
  const int col   = (wave << 4) + (lane & 15);
  const int mhalf = (lane >> 4) << 3;

  int tile = blockIdx.x;
  const int stride = gridDim.x;

  // prologue: tile t -> bf16 buf0 ; async copy of tile t+1 in flight
#if USE_ASYNC_STAGE
  if (tile < M_TILES) {
    stage_tile_async(sF, x + (size_t)tile * (16 * INSZ), tid);
    wait_async0();
    convert_tile(sB, sF, tid);
    if (tile + stride < M_TILES)
      stage_tile_async(sF, x + (size_t)(tile + stride) * (16 * INSZ), tid);
  }
#else
  if (tile < M_TILES)
    stage_tile(sB, x + (size_t)tile * (16 * INSZ), tid);
#endif
  __syncthreads();

  for (int it = 0; tile < M_TILES; tile += stride, ++it) {
    const __bf16* cur = sB + ((it & 1) << 12);
    __bf16*       nxt = sB + (((it + 1) & 1) << 12);

#if !USE_ASYNC_STAGE
    if (tile + stride < M_TILES)
      stage_tile(nxt, x + (size_t)(tile + stride) * (16 * INSZ), tid);
#endif

    v8f acc = {};
    v16bf a_cur = load_a_frag(cur, lane, 0);
#pragma unroll
    for (int kb = 0; kb < 8; ++kb) {
      v16bf a_nxt;
      if (kb < 7) a_nxt = load_a_frag(cur, lane, kb + 1);
      acc = __builtin_amdgcn_wmma_f32_16x16x32_bf16(false, a_cur, false, ufrag[kb],
                                                    (short)0, acc, false, false);
      a_cur = a_nxt;
    }

    float* obase = dec + (size_t)tile * 16 * HID;
#pragma unroll
    for (int r = 0; r < 8; ++r)
      obase[(size_t)(mhalf + r) * HID + col] = acc[r];

#if USE_ASYNC_STAGE
    if (tile + stride < M_TILES) {
      wait_async0();                          // tile t+1 landed in sF
      convert_tile(nxt, sF, tid);             // thread-private range, no barrier
      if (tile + 2 * stride < M_TILES)        // async engine fetches tile t+2
        stage_tile_async(sF, x + (size_t)(tile + 2 * stride) * (16 * INSZ), tid);
    }
#endif
    __syncthreads();                          // next iteration flips buffers
  }
}

// ------------------------------ phase 2: recurrence h = tanh(XU + h@V + b) --
// 16 workgroups x 512 threads; WG g owns batch rows [g*16, g*16+16).
// V fragments live in registers for all 512 steps; LDS holds only the
// double-buffered 16x256 bf16 h tile (16 KB). XU is prefetched one step ahead.
__global__ void rnn_rec_kernel(const float* __restrict__ enc,  // [BZ, HID]
                               const float* __restrict__ V,    // [HID, HID]
                               const float* __restrict__ bias, // [HID]
                               float* __restrict__ out) {
  extern __shared__ char smem[];
  __bf16* sH = (__bf16*)smem;                 // 2 x 8192 B

  const int tid  = threadIdx.x;
  const int lane = tid & 31;
  const int wave = tid >> 5;
  const int g    = blockIdx.x;                // batch-row block

  // register-resident V fragments (invariant per wave across all timesteps)
  v16bf vfrag[8];
#pragma unroll
  for (int kb = 0; kb < 8; ++kb)
    vfrag[kb] = load_w_frag_global(V, lane, kb, wave);

  // init h from encoder_inputs (vectorized) + produce the enc output copy
  {
    const int i = tid << 3;                   // element in the 16x256 block
    const int m = i >> 8, n = i & 255;
    const float* src = enc + ((g << 4) + m) * HID + n;
    v4f a = *(const v4f*)src;
    v4f b = *(const v4f*)(src + 4);
    float* dst = out + ((g << 4) + m) * HID + n;
    *(v4f*)dst       = a;
    *(v4f*)(dst + 4) = b;
    *(v8bf*)(sH + i) = cvt8(a, b);            // buffer 0
  }
  __syncthreads();

  const int   col   = (wave << 4) + (lane & 15);
  const int   mhalf = (lane >> 4) << 3;
  const float bval  = bias[col];
  float* dec = out + BZ * HID;

  // software-pipelined XU fragment (one step ahead)
  float xu[8];
#pragma unroll
  for (int r = 0; r < 8; ++r) {
    const int b = (g << 4) + mhalf + r;
    xu[r] = dec[((size_t)b * SEQ + 0) * HID + col];
  }

  for (int t = 0; t < SEQ; ++t) {
    const __bf16* hcur = sH + ((t & 1) << 12);
    __bf16*       hnxt = sH + (((t + 1) & 1) << 12);

    v8f acc;
#pragma unroll
    for (int r = 0; r < 8; ++r) acc[r] = xu[r] + bval;

    if (t + 1 < SEQ) {                        // issue next step's loads early
#pragma unroll
      for (int r = 0; r < 8; ++r) {
        const int b = (g << 4) + mhalf + r;
        xu[r] = dec[((size_t)b * SEQ + (t + 1)) * HID + col];
      }
    }

    v16bf a_cur = load_a_frag(hcur, lane, 0);
#pragma unroll
    for (int kb = 0; kb < 8; ++kb) {
      v16bf a_nxt;
      if (kb < 7) a_nxt = load_a_frag(hcur, lane, kb + 1);
      acc = __builtin_amdgcn_wmma_f32_16x16x32_bf16(false, a_cur, false, vfrag[kb],
                                                    (short)0, acc, false, false);
      a_cur = a_nxt;
    }

#pragma unroll
    for (int r = 0; r < 8; ++r) {
      const float hv = fast_tanh(acc[r]);
      const int b = (g << 4) + mhalf + r;
      hnxt[(mhalf + r) * 256 + col] = (__bf16)hv;        // A operand for t+1
      dec[((size_t)b * SEQ + t) * HID + col] = hv;       // overwrite XU slot
    }
    __syncthreads();   // hnxt complete before all waves read it next step
  }
}

// --------------------------------------------------------------- launcher ---
extern "C" void kernel_launch(void* const* d_in, const int* in_sizes, int n_in,
                              void* d_out, int out_size, void* d_ws, size_t ws_size,
                              hipStream_t stream) {
  (void)in_sizes; (void)n_in; (void)out_size; (void)d_ws; (void)ws_size;
  const float* enc = (const float*)d_in[0];   // [BZ, HID]
  const float* x   = (const float*)d_in[1];   // [BZ, SEQ, INSZ]
  const float* U   = (const float*)d_in[2];   // [INSZ, HID]
  const float* V   = (const float*)d_in[3];   // [HID, HID]
  const float* b   = (const float*)d_in[4];   // [HID]
  float* out = (float*)d_out;

  const size_t smem1 = 32768;                 // f32 staging + 2 bf16 tiles
  const size_t smem2 = 16384;                 // double-buffered h

  rnn_xu_kernel<<<1024, 512, smem1, stream>>>(x, U, out);
  rnn_rec_kernel<<<16, 512, smem2, stream>>>(enc, V, b, out);
}
